// PPNRelationNet_21706764714081
// MI455X (gfx1250) — compile-verified, matrix operand order checked by reading, not run on hardware
//
#include <hip/hip_runtime.h>
#include <hip/hip_bf16.h>
#include <math.h>

// ---------------------------------------------------------------------------
// Problem constants (from reference)
// ---------------------------------------------------------------------------
#define BATCH     256
#define CLS       500
#define CLSP      512          // CLS padded to multiple of 32
#define ATT_DIM   1024
#define IMG_DIM   1024
#define ATT_HC    256
#define HIDDEN_C  1024
#define T_TEMP    10.0f
#define THRESH    0.86602540378443864676f   // cos(30 deg)
#define NEGVAL    (-9.0e15f)

typedef __attribute__((ext_vector_type(16))) __bf16 v16bf;
typedef __attribute__((ext_vector_type(8)))  __bf16 v8bf;
typedef __attribute__((ext_vector_type(8)))  float  v8f;
typedef __attribute__((ext_vector_type(4)))  unsigned int u32x4;
typedef __attribute__((ext_vector_type(8)))  int  i32x8;
typedef __attribute__((ext_vector_type(4)))  int  i32x4;

// ---------------------------------------------------------------------------
// f32 -> bf16 convert with zero row padding (row-major copy)
// ---------------------------------------------------------------------------
__global__ __launch_bounds__(256)
void cvt_pad_bf16(const float* __restrict__ src, __bf16* __restrict__ dst,
                  int srcRows, int dstRows, int cols) {
    size_t i   = (size_t)blockIdx.x * blockDim.x + threadIdx.x;
    size_t tot = (size_t)dstRows * cols;
    if (i >= tot) return;
    int r = (int)(i / cols);
    dst[i] = (r < srcRows) ? (__bf16)src[i] : (__bf16)0.0f;
}

// ---------------------------------------------------------------------------
// f32 [R,C] -> bf16 transposed [C,Rpad] (zero pad along K)
// ---------------------------------------------------------------------------
__global__ __launch_bounds__(256)
void transpose_cvt_bf16(const float* __restrict__ src, __bf16* __restrict__ dst,
                        int R, int C, int Rpad) {
    size_t i   = (size_t)blockIdx.x * blockDim.x + threadIdx.x;
    size_t tot = (size_t)C * Rpad;
    if (i >= tot) return;
    int c = (int)(i / Rpad);
    int k = (int)(i % Rpad);
    dst[i] = (k < R) ? (__bf16)src[(size_t)k * C + c] : (__bf16)0.0f;
}

// ---------------------------------------------------------------------------
// WMMA fragment loader (CDNA5 ISA 7.12.2, 16-bit A/B 16x32 layout):
//   lanes 0-15  : row r = lane,    K-chunks [k0, k0+8) and [k0+16, k0+24)
//   lanes 16-31 : row r = lane-16, K-chunks [k0+8,k0+16) and [k0+24, k0+32)
// rowbase already includes (row * K + (lane>>4)*8).
// ---------------------------------------------------------------------------
__device__ __forceinline__ v16bf load_frag(const __bf16* rowbase, int k0) {
    v8bf x0 = *(const v8bf*)(rowbase + k0);
    v8bf x1 = *(const v8bf*)(rowbase + k0 + 16);
    return __builtin_shufflevector(x0, x1, 0,1,2,3,4,5,6,7,8,9,10,11,12,13,14,15);
}

__device__ __forceinline__ v8f wmma_bf16(v16bf a, v16bf b, v8f c) {
    return __builtin_amdgcn_wmma_f32_16x16x32_bf16(false, a, false, b,
                                                   (short)0, c, false, false);
}

// ---------------------------------------------------------------------------
// Generic WMMA GEMM:  C[M,N] = A[M,K] * B[K,N]
//   A  : bf16 row-major [M,K];  Bt : bf16 N-major [N,K] (pre-transposed B)
//   C  : f32 or bf16 row-major [M,N]
// One wave32 per 32x32 output tile: 2 A frags x 2 B frags -> 4 independent
// v_wmma_f32_16x16x32_bf16 accumulator chains per K-step (keeps XDL pipe full,
// halves loads per WMMA vs a 16x16 tile). M,N multiples of 32; K multiple of 32.
// ---------------------------------------------------------------------------
template <bool OUT_BF16>
__global__ __launch_bounds__(256)
void gemm_wmma_bf16(const __bf16* __restrict__ A, const __bf16* __restrict__ Bt,
                    void* __restrict__ Cout, int M, int N, int K) {
    const int lane   = threadIdx.x;            // 0..31
    const int tilesN = N >> 5;
    const int tile   = blockIdx.x * blockDim.y + threadIdx.y;  // wave-uniform
    const int tm     = (tile / tilesN) << 5;
    const int tn     = (tile % tilesN) << 5;

    const int r  = lane & 15;
    const int hi = lane >> 4;                  // 0 or 1 (lane half)

    const __bf16* a0p = A  + (size_t)(tm +      r) * K + hi * 8;
    const __bf16* a1p = A  + (size_t)(tm + 16 + r) * K + hi * 8;
    const __bf16* b0p = Bt + (size_t)(tn +      r) * K + hi * 8;
    const __bf16* b1p = Bt + (size_t)(tn + 16 + r) * K + hi * 8;

    v8f c00 = {}, c01 = {}, c10 = {}, c11 = {};
    for (int k0 = 0; k0 < K; k0 += 32) {
        if (k0 + 256 < K) {                    // uniform; L2 prefetch 1 block ahead
            __builtin_prefetch(a0p + k0 + 256, 0, 1);
            __builtin_prefetch(b0p + k0 + 256, 0, 1);
        }
        v16bf a0 = load_frag(a0p, k0);
        v16bf a1 = load_frag(a1p, k0);
        v16bf b0 = load_frag(b0p, k0);
        v16bf b1 = load_frag(b1p, k0);
        c00 = wmma_bf16(a0, b0, c00);
        c01 = wmma_bf16(a0, b1, c01);
        c10 = wmma_bf16(a1, b0, c10);
        c11 = wmma_bf16(a1, b1, c11);
    }

    // C/D layout: lane holds column n = (lane&15); VGPR i holds row i + 8*(lane>>4).
    #pragma unroll
    for (int i = 0; i < 8; ++i) {
        const int m0 = tm + i + hi * 8;
        const int m1 = m0 + 16;
        const int n0 = tn + r;
        const int n1 = n0 + 16;
        if (OUT_BF16) {
            __bf16* C = (__bf16*)Cout;
            C[(size_t)m0 * N + n0] = (__bf16)c00[i];
            C[(size_t)m0 * N + n1] = (__bf16)c01[i];
            C[(size_t)m1 * N + n0] = (__bf16)c10[i];
            C[(size_t)m1 * N + n1] = (__bf16)c11[i];
        } else {
            float* C = (float*)Cout;
            C[(size_t)m0 * N + n0] = c00[i];
            C[(size_t)m0 * N + n1] = c01[i];
            C[(size_t)m1 * N + n0] = c10[i];
            C[(size_t)m1 * N + n1] = c11[i];
        }
    }
}

// ---------------------------------------------------------------------------
// Row L2-normalize f32 [rows,cols] -> bf16 [rows,cols]; one wave32 per row.
// ---------------------------------------------------------------------------
__global__ __launch_bounds__(32)
void rownorm_to_bf16(const float* __restrict__ g, __bf16* __restrict__ gn, int cols) {
    const int row  = blockIdx.x;
    const int lane = threadIdx.x;
    const float* gr = g + (size_t)row * cols;
    float s = 0.0f;
    for (int i = lane; i < cols; i += 32) { float v = gr[i]; s += v * v; }
    #pragma unroll
    for (int off = 16; off > 0; off >>= 1) s += __shfl_xor(s, off, 32);
    const float rinv = rsqrtf(s + 1e-24f);     // padded zero rows stay zero
    __bf16* dr = gn + (size_t)row * cols;
    for (int i = lane; i < cols; i += 32) dr[i] = (__bf16)(gr[i] * rinv);
}

// ---------------------------------------------------------------------------
// Masked, temperature-scaled softmax over each row of sim [n,n] -> bf16 attn.
// One wave32 per row. mask: sim > THRESH ? sim : NEG, then *T_TEMP.
// Padded columns (zero gn rows) fall below THRESH -> exp(-9e16) -> 0.
// ---------------------------------------------------------------------------
__global__ __launch_bounds__(32)
void masked_softmax_bf16(const float* __restrict__ sim, __bf16* __restrict__ attn, int n) {
    const int row  = blockIdx.x;
    const int lane = threadIdx.x;
    const float* sr = sim + (size_t)row * n;

    float mx = -3.0e38f;
    for (int i = lane; i < n; i += 32) {
        float v = sr[i];
        float x = ((v > THRESH) ? v : NEGVAL) * T_TEMP;
        mx = fmaxf(mx, x);
    }
    #pragma unroll
    for (int off = 16; off > 0; off >>= 1) mx = fmaxf(mx, __shfl_xor(mx, off, 32));

    float sum = 0.0f;
    for (int i = lane; i < n; i += 32) {
        float v = sr[i];
        float x = ((v > THRESH) ? v : NEGVAL) * T_TEMP;
        sum += __expf(x - mx);
    }
    #pragma unroll
    for (int off = 16; off > 0; off >>= 1) sum += __shfl_xor(sum, off, 32);

    const float inv = 1.0f / sum;
    __bf16* ar = attn + (size_t)row * n;
    for (int i = lane; i < n; i += 32) {
        float v = sr[i];
        float x = ((v > THRESH) ? v : NEGVAL) * T_TEMP;
        ar[i] = (__bf16)(__expf(x - mx) * inv);
    }
}

// ---------------------------------------------------------------------------
// Tensor Data Mover: 2-D f32 tile load Global -> LDS (ISA ch.8 D# layout).
// group0: count=1 | lds_addr | global_addr | type=2
// group1: data_size=4B, tensor_dim0/1, tile_dim0/1, tensor_dim0_stride
// Issued once per wave (TDM ignores EXEC); tracked by TENSORcnt.
// ---------------------------------------------------------------------------
#if __has_builtin(__builtin_amdgcn_tensor_load_to_lds)
#define HAVE_TDM 1
__device__ __forceinline__ void tdm_load_2d_f32(unsigned int lds_off,
                                                const float* gptr,
                                                unsigned int tensor_w,
                                                unsigned int tensor_h,
                                                unsigned int tile_w,
                                                unsigned int tile_h,
                                                unsigned int row_stride_elems) {
    unsigned long long ga = (unsigned long long)(uintptr_t)gptr;
    u32x4 g0;
    g0[0] = 1u;                                         // count=1 (valid user D#)
    g0[1] = lds_off;                                    // lds_addr (bytes)
    g0[2] = (unsigned int)(ga & 0xffffffffu);           // global_addr[31:0]
    g0[3] = (unsigned int)((ga >> 32) & 0x01ffffffu)    // global_addr[56:32]
          | (2u << 30);                                 // type=2 ("image")
    i32x8 g1;
    g1[0] = (int)(2u << 16);                            // data_size=2 -> 4 bytes
    g1[1] = (int)((tensor_w & 0xffffu) << 16);          // tensor_dim0[15:0]
    g1[2] = (int)((tensor_w >> 16) | ((tensor_h & 0xffffu) << 16));
    g1[3] = (int)((tensor_h >> 16) | ((tile_w & 0xffffu) << 16));   // tile_dim0
    g1[4] = (int)(tile_h & 0xffffu);                    // tile_dim1 (tile_dim2=0)
    g1[5] = (int)row_stride_elems;                      // tensor_dim0_stride[31:0]
    g1[6] = 0;
    g1[7] = 0;
    i32x4 z4 = {0, 0, 0, 0};
#if defined(__clang_major__) && (__clang_major__ >= 23)
    i32x8 z8 = {0, 0, 0, 0, 0, 0, 0, 0};
    __builtin_amdgcn_tensor_load_to_lds(g0, g1, z4, z4, z8, 0);
#else
    __builtin_amdgcn_tensor_load_to_lds(g0, g1, z4, z4, 0);
#endif
}
#else
#define HAVE_TDM 0
#endif

__device__ __forceinline__ unsigned int lds_offset_of(const void* p) {
    // generic pointer into LDS: addr[31:0] is the LDS byte offset (ISA 10.2)
    return (unsigned int)(uintptr_t)p;
}

// ---------------------------------------------------------------------------
// Fused epilogue: out[b,c] = sum_h relu(img_h[b,h]+sem_h[c,h]+sem_b[h])*fc_w[h] + fc_b
// Never materializes the 256x500x1024 "hidden" tensor (would be 524 MB).
// 256-thread block -> 16x16 output tile; H tiled by 128. The two 16x128 f32
// tiles are DMA'd into LDS by the Tensor Data Mover (wave 0 issues, waits on
// TENSORcnt, workgroup barrier publishes).
// ---------------------------------------------------------------------------
#define HTILE 128
__global__ __launch_bounds__(256)
void fused_relu_dot(const float* __restrict__ img_h, const float* __restrict__ sem_h,
                    const float* __restrict__ sem_b, const float* __restrict__ fc_w,
                    const float* __restrict__ fc_b, float* __restrict__ out) {
    __shared__ float sImg[16][HTILE];
    __shared__ float sSem[16][HTILE];
    __shared__ float sB[HTILE];
    __shared__ float sW[HTILE];

    const int t  = threadIdx.x;     // 0..255
    const int bi = t >> 4;          // 0..15 (batch within tile)
    const int ci = t & 15;          // 0..15 (class within tile)
    const int b0 = blockIdx.x * 16;
    const int c0 = blockIdx.y * 16;

    float acc = 0.0f;
    for (int h0 = 0; h0 < HIDDEN_C; h0 += HTILE) {
#if HAVE_TDM
        if (t < 32) {               // wave 0 only; TDM issues once per wave
            tdm_load_2d_f32(lds_offset_of(&sImg[0][0]),
                            img_h + (size_t)b0 * HIDDEN_C + h0,
                            HIDDEN_C, 16, HTILE, 16, HIDDEN_C);
            tdm_load_2d_f32(lds_offset_of(&sSem[0][0]),
                            sem_h + (size_t)c0 * HIDDEN_C + h0,
                            HIDDEN_C, 16, HTILE, 16, HIDDEN_C);
            __builtin_amdgcn_s_wait_tensorcnt(0);
        }
#else
        for (int i = t; i < 16 * HTILE; i += 256) {
            int rr = i / HTILE, cc = i % HTILE;
            sImg[rr][cc] = img_h[(size_t)(b0 + rr) * HIDDEN_C + h0 + cc];
            sSem[rr][cc] = sem_h[(size_t)(c0 + rr) * HIDDEN_C + h0 + cc];
        }
#endif
        for (int i = t; i < HTILE; i += 256) {
            sB[i] = sem_b[h0 + i];
            sW[i] = fc_w[h0 + i];
        }
        __syncthreads();

        const float* ir = sImg[bi];
        const float* cr = sSem[ci];
        float a = 0.0f;
        #pragma unroll 8
        for (int h = 0; h < HTILE; ++h) {
            float v = ir[h] + cr[h] + sB[h];
            a += fmaxf(v, 0.0f) * sW[h];
        }
        acc += a;
        __syncthreads();
    }

    const int c = c0 + ci;
    if (c < CLS) out[(size_t)(b0 + bi) * CLS + c] = acc + fc_b[0];
}

// ---------------------------------------------------------------------------
// Host launcher
// ---------------------------------------------------------------------------
static inline size_t align256(size_t x) { return (x + 255u) & ~(size_t)255u; }

extern "C" void kernel_launch(void* const* d_in, const int* in_sizes, int n_in,
                              void* d_out, int out_size, void* d_ws, size_t ws_size,
                              hipStream_t stream) {
    (void)in_sizes; (void)n_in; (void)out_size; (void)ws_size;

    const float* image_feats = (const float*)d_in[0];   // [256,1024]
    const float* attributes  = (const float*)d_in[1];   // [500,1024]
    //            d_in[2] unused scalar
    const float* att_g       = (const float*)d_in[3];   // [1024,256]
    const float* img_w       = (const float*)d_in[4];   // [1024,1024]
    const float* sem_w       = (const float*)d_in[5];   // [1024,1024]
    const float* sem_b       = (const float*)d_in[6];   // [1,1024]
    const float* fc_w        = (const float*)d_in[7];   // [1,1024]
    const float* fc_b        = (const float*)d_in[8];   // [1]
    float*       out         = (float*)d_out;           // [256,500]

    // ---- workspace carve-up ------------------------------------------------
    char* p = (char*)d_ws;
    size_t off = 0;
    auto take = [&](size_t bytes) { char* q = p + off; off = align256(off + bytes); return q; };

    __bf16* attrs_bf   = (__bf16*)take((size_t)CLSP * ATT_DIM * 2);     // A: padded attributes
    __bf16* attT_bf    = (__bf16*)take((size_t)ATT_DIM * CLSP * 2);     // attributes^T (N-major)
    __bf16* attgT_bf   = (__bf16*)take((size_t)ATT_HC * ATT_DIM * 2);   // att_g^T
    __bf16* imgwT_bf   = (__bf16*)take((size_t)HIDDEN_C * IMG_DIM * 2); // img_w^T
    __bf16* semwT_bf   = (__bf16*)take((size_t)HIDDEN_C * ATT_DIM * 2); // sem_w^T
    __bf16* imgf_bf    = (__bf16*)take((size_t)BATCH * IMG_DIM * 2);    // image_feats
    float*  g_f32      = (float*)take((size_t)CLSP * ATT_HC * 4);       // attributes@att_g
    __bf16* gn_bf      = (__bf16*)take((size_t)CLSP * ATT_HC * 2);      // row-normalized
    float*  sim_f32    = (float*)take((size_t)CLSP * CLSP * 4);         // gn@gn^T
    __bf16* attn_bf    = (__bf16*)take((size_t)CLSP * CLSP * 2);        // softmax
    __bf16* attouts_bf = (__bf16*)take((size_t)CLSP * ATT_DIM * 2);     // attention@attributes
    float*  imgh_f32   = (float*)take((size_t)BATCH * HIDDEN_C * 4);    // image_feats@img_w
    float*  semh_f32   = (float*)take((size_t)CLSP * HIDDEN_C * 4);     // att_outs@sem_w

    // ---- stage 0: convert / transpose to bf16 ------------------------------
    {
        size_t n;
        n = (size_t)CLSP * ATT_DIM;
        cvt_pad_bf16<<<(unsigned)((n + 255) / 256), 256, 0, stream>>>(attributes, attrs_bf, CLS, CLSP, ATT_DIM);
        n = (size_t)BATCH * IMG_DIM;
        cvt_pad_bf16<<<(unsigned)((n + 255) / 256), 256, 0, stream>>>(image_feats, imgf_bf, BATCH, BATCH, IMG_DIM);
        n = (size_t)ATT_DIM * CLSP;
        transpose_cvt_bf16<<<(unsigned)((n + 255) / 256), 256, 0, stream>>>(attributes, attT_bf, CLS, ATT_DIM, CLSP);
        n = (size_t)ATT_HC * ATT_DIM;
        transpose_cvt_bf16<<<(unsigned)((n + 255) / 256), 256, 0, stream>>>(att_g, attgT_bf, ATT_DIM, ATT_HC, ATT_DIM);
        n = (size_t)HIDDEN_C * IMG_DIM;
        transpose_cvt_bf16<<<(unsigned)((n + 255) / 256), 256, 0, stream>>>(img_w, imgwT_bf, IMG_DIM, HIDDEN_C, IMG_DIM);
        n = (size_t)HIDDEN_C * ATT_DIM;
        transpose_cvt_bf16<<<(unsigned)((n + 255) / 256), 256, 0, stream>>>(sem_w, semwT_bf, ATT_DIM, HIDDEN_C, ATT_DIM);
    }

    const dim3 gblk(32, 8);   // 8 wave32 per block, one 32x32 tile each
    auto gemm_grid = [](int M, int N) { return (unsigned)(((M >> 5) * (N >> 5)) / 8); };

    // ---- stage 1: g = attributes @ att_g  [512,256] ------------------------
    gemm_wmma_bf16<false><<<gemm_grid(CLSP, ATT_HC), gblk, 0, stream>>>(
        attrs_bf, attgT_bf, g_f32, CLSP, ATT_HC, ATT_DIM);

    // ---- stage 2: gn = row-normalize(g) ------------------------------------
    rownorm_to_bf16<<<CLSP, 32, 0, stream>>>(g_f32, gn_bf, ATT_HC);

    // ---- stage 3: sim = gn @ gn^T  [512,512] (Bt = gn itself) --------------
    gemm_wmma_bf16<false><<<gemm_grid(CLSP, CLSP), gblk, 0, stream>>>(
        gn_bf, gn_bf, sim_f32, CLSP, CLSP, ATT_HC);

    // ---- stage 4: attention = softmax(mask(sim)*T) -------------------------
    masked_softmax_bf16<<<CLSP, 32, 0, stream>>>(sim_f32, attn_bf, CLSP);

    // ---- stage 5: att_outs = attention @ attributes  [512,1024] bf16 -------
    gemm_wmma_bf16<true><<<gemm_grid(CLSP, ATT_DIM), gblk, 0, stream>>>(
        attn_bf, attT_bf, attouts_bf, CLSP, ATT_DIM, CLSP);

    // ---- stage 6: img_h = image_feats @ img_w  [256,1024] ------------------
    gemm_wmma_bf16<false><<<gemm_grid(BATCH, HIDDEN_C), gblk, 0, stream>>>(
        imgf_bf, imgwT_bf, imgh_f32, BATCH, HIDDEN_C, IMG_DIM);

    // ---- stage 7: sem_h = att_outs @ sem_w  [512,1024] ---------------------
    gemm_wmma_bf16<false><<<gemm_grid(CLSP, HIDDEN_C), gblk, 0, stream>>>(
        attouts_bf, semwT_bf, semh_f32, CLSP, HIDDEN_C, ATT_DIM);

    // ---- stage 8: fused relu-dot epilogue -> out [256,500] -----------------
    {
        dim3 grid(BATCH / 16, CLSP / 16);
        fused_relu_dot<<<grid, 256, 0, stream>>>(imgh_f32, semh_f32, sem_b, fc_w, fc_b, out);
    }
}
